// SlotAttention_41059887350440
// MI455X (gfx1250) — matrix operand order repaired and checked by reading, not compiled, not run on hardware
//
#include <hip/hip_runtime.h>

#define D 256
#define NTOK 4096
#define NB 32
#define NSLOT 8
#define N_ITERS 3

typedef __attribute__((ext_vector_type(16))) _Float16 v16h;
typedef __attribute__((ext_vector_type(8)))  _Float16 h8v;
typedef __attribute__((ext_vector_type(8)))  float    v8f;

union H16 {
  v16h v;
  h8v  h[2];
  _Float16 e[16];
};

__device__ __forceinline__ v8f wmma_f16(v16h a, v16h b, v8f c) {
  // 8 args: (neg_a, A, neg_b, B, c_mod, C, reuse_a, reuse_b)
  return __builtin_amdgcn_wmma_f32_16x16x32_f16(false, a, false, b, (short)0, c, false, false);
}

// A-matrix tile (16x32 f16), row-major source with stride ldm (halves).
// Layout: lane holds row M = lane&15; halves h -> K = kb + h + (h&8), kb = (lane>>4)*8.
__device__ __forceinline__ v16h load_A(const _Float16* src, int ldm, int lane) {
  const int row = lane & 15;
  const int kb  = (lane >> 4) << 3;
  const _Float16* p = src + row * ldm + kb;
  H16 u;
  u.h[0] = *(const h8v*)(p);        // K = kb..kb+7
  u.h[1] = *(const h8v*)(p + 16);   // K = kb+16..kb+23
  return u.v;
}

// B-matrix tile (32x16 f16), row-major source (K rows x N cols), stride ld (halves).
// Layout: lane holds column N = lane&15; halves h -> K = kb + h + (h&8), kb = (lane>>4)*8.
__device__ __forceinline__ v16h load_B(const _Float16* src, long ld, int lane) {
  const _Float16* p = src + (lane & 15) + (long)((lane >> 4) << 3) * ld;
  H16 u;
  #pragma unroll
  for (int s = 0; s < 16; ++s) u.e[s] = p[(long)(s + (s & 8)) * ld];
  return u.v;
}

// ---------------------------------------------------------------------------
// Weight fp32 -> fp16 conversion (done once; weights then L2-resident as f16)
// ---------------------------------------------------------------------------
__global__ void convert_w_kernel(const float* __restrict__ wk, const float* __restrict__ wv,
                                 _Float16* __restrict__ wk16, _Float16* __restrict__ wv16) {
  const int i = blockIdx.x * blockDim.x + threadIdx.x;
  if (i < D * D) {
    wk16[i] = (_Float16)wk[i];
    wv16[i] = (_Float16)wv[i];
  }
}

__global__ void init_slots_kernel(const float* __restrict__ emb, float* __restrict__ slots) {
  const int b = blockIdx.x;
  for (int i = threadIdx.x; i < NSLOT * D; i += blockDim.x)
    slots[(long)b * NSLOT * D + i] = emb[i];
}

// ---------------------------------------------------------------------------
// Fused LayerNorm(inputs) + K/V projection GEMMs (the 34-GFLOP hot kernel).
// K stored transposed [b][d][token] f16 (so dots-B tiles are row-major-in-K),
// V stored row-major  [b][token][d] f16 (so updates-B tiles are row-major-in-K).
// Block = 128 threads (4 waves), 64 rows per block, 16-row M-tile per wave.
// ---------------------------------------------------------------------------
__global__ void __launch_bounds__(128)
kv_proj_kernel(const float* __restrict__ inp,
               const float* __restrict__ g, const float* __restrict__ bvec,
               const _Float16* __restrict__ wk16, const _Float16* __restrict__ wv16,
               const float* __restrict__ bk, const float* __restrict__ bv,
               _Float16* __restrict__ kt, _Float16* __restrict__ vout) {
  const int lane = threadIdx.x & 31;
  const int wave = threadIdx.x >> 5;          // 0..3
  const long row0 = (long)blockIdx.x * 64;
  __shared__ __align__(16) _Float16 x16[64 * 264];

  const float4 g0  = *(const float4*)(g + lane * 8);
  const float4 g1  = *(const float4*)(g + lane * 8 + 4);
  const float4 lb0 = *(const float4*)(bvec + lane * 8);
  const float4 lb1 = *(const float4*)(bvec + lane * 8 + 4);

  // Phase A: LayerNorm 16 rows per wave -> f16 in LDS
  for (int r = wave * 16; r < wave * 16 + 16; ++r) {
    const float* src = inp + (row0 + r) * D + lane * 8;
    float4 a = *(const float4*)(src);
    float4 c = *(const float4*)(src + 4);
    float s  = a.x + a.y + a.z + a.w + c.x + c.y + c.z + c.w;
    float ss = a.x*a.x + a.y*a.y + a.z*a.z + a.w*a.w
             + c.x*c.x + c.y*c.y + c.z*c.z + c.w*c.w;
    #pragma unroll
    for (int m = 16; m; m >>= 1) { s += __shfl_xor(s, m, 32); ss += __shfl_xor(ss, m, 32); }
    const float mean = s * (1.f / 256.f);
    const float inv  = rsqrtf(ss * (1.f / 256.f) - mean * mean + 1e-5f);
    _Float16* dst = x16 + r * 264 + lane * 8;
    dst[0] = (_Float16)((a.x - mean) * inv * g0.x + lb0.x);
    dst[1] = (_Float16)((a.y - mean) * inv * g0.y + lb0.y);
    dst[2] = (_Float16)((a.z - mean) * inv * g0.z + lb0.z);
    dst[3] = (_Float16)((a.w - mean) * inv * g0.w + lb0.w);
    dst[4] = (_Float16)((c.x - mean) * inv * g1.x + lb1.x);
    dst[5] = (_Float16)((c.y - mean) * inv * g1.y + lb1.y);
    dst[6] = (_Float16)((c.z - mean) * inv * g1.z + lb1.z);
    dst[7] = (_Float16)((c.w - mean) * inv * g1.w + lb1.w);
  }
  __syncthreads();

  // Phase B: per wave, 16 rows x 256 cols for Wk then Wv (WMMA f32<-f16)
  const long wrow = row0 + wave * 16;
  const int  m_hi = (lane >> 4) << 3;
  const int  ncol = lane & 15;
  for (int mat = 0; mat < 2; ++mat) {
    const _Float16* W   = mat ? wv16 : wk16;
    const float*    bia = mat ? bv : bk;
    for (int nt = 0; nt < 16; ++nt) {
      v8f acc = {};
      #pragma unroll
      for (int ks = 0; ks < 8; ++ks) {
        v16h A  = load_A(x16 + wave * 16 * 264 + ks * 32, 264, lane);
        v16h Bm = load_B(W + (ks * 32) * D + nt * 16, D, lane);
        acc = wmma_f16(A, Bm, acc);
      }
      const int col = nt * 16 + ncol;
      const float bcol = bia[col];
      #pragma unroll
      for (int r = 0; r < 8; ++r) {
        const float val = acc[r] + bcol;       // C tile: vgpr r = row (m&7), lane%16 = col
        const long grow = wrow + r + m_hi;
        if (mat == 0) {
          const long bidx = grow >> 12;        // /4096
          const long tok  = grow & 4095;
          kt[(bidx * D + col) * (long)NTOK + tok] = (_Float16)val;
        } else {
          vout[grow * D + col] = (_Float16)val;
        }
      }
    }
  }
}

// ---------------------------------------------------------------------------
// q = (LN(slots) @ Wq + bq) * d^-0.5, stored f16. Tiny; one block per batch,
// one wave per slot row.
// ---------------------------------------------------------------------------
__global__ void __launch_bounds__(256)
q_proj_kernel(const float* __restrict__ slots,
              const float* __restrict__ g, const float* __restrict__ bvec,
              const float* __restrict__ Wq, const float* __restrict__ bq,
              _Float16* __restrict__ q16) {
  const int b = blockIdx.x;
  const int lane = threadIdx.x & 31;
  const int wave = threadIdx.x >> 5;          // slot row 0..7
  __shared__ float srow[NSLOT][D];
  const float* sp = slots + ((long)b * NSLOT + wave) * D + lane * 8;
  float4 a = *(const float4*)(sp);
  float4 c = *(const float4*)(sp + 4);
  float s  = a.x + a.y + a.z + a.w + c.x + c.y + c.z + c.w;
  float ss = a.x*a.x + a.y*a.y + a.z*a.z + a.w*a.w
           + c.x*c.x + c.y*c.y + c.z*c.z + c.w*c.w;
  #pragma unroll
  for (int m = 16; m; m >>= 1) { s += __shfl_xor(s, m, 32); ss += __shfl_xor(ss, m, 32); }
  const float mean = s * (1.f / 256.f);
  const float inv  = rsqrtf(ss * (1.f / 256.f) - mean * mean + 1e-5f);
  const float4 g0  = *(const float4*)(g + lane * 8);
  const float4 g1  = *(const float4*)(g + lane * 8 + 4);
  const float4 n0  = *(const float4*)(bvec + lane * 8);
  const float4 n1  = *(const float4*)(bvec + lane * 8 + 4);
  float* dst = &srow[wave][lane * 8];
  dst[0] = (a.x - mean) * inv * g0.x + n0.x;
  dst[1] = (a.y - mean) * inv * g0.y + n0.y;
  dst[2] = (a.z - mean) * inv * g0.z + n0.z;
  dst[3] = (a.w - mean) * inv * g0.w + n0.w;
  dst[4] = (c.x - mean) * inv * g1.x + n1.x;
  dst[5] = (c.y - mean) * inv * g1.y + n1.y;
  dst[6] = (c.z - mean) * inv * g1.z + n1.z;
  dst[7] = (c.w - mean) * inv * g1.w + n1.w;
  __syncthreads();
  for (int j = 0; j < 8; ++j) {
    const int col = j * 32 + lane;
    float acc = 0.f;
    for (int i = 0; i < D; ++i) acc = fmaf(srow[wave][i], Wq[i * D + col], acc);
    acc = (acc + bq[col]) * 0.0625f;          // * d^-0.5
    q16[((long)b * NSLOT + wave) * D + col] = (_Float16)acc;
  }
}

// ---------------------------------------------------------------------------
// Attention: per (batch, 256-token chunk). Per wave: dots (2 N-tiles x 8
// k-steps WMMA), per-lane softmax over 8 slots (+EPS), LDS transpose of attn,
// then updates (2 d-tiles x 8 k-steps WMMA) atomically accumulated in f32.
// ---------------------------------------------------------------------------
__global__ void __launch_bounds__(256)
attn_kernel(const _Float16* __restrict__ q16,
            const _Float16* __restrict__ kt,
            const _Float16* __restrict__ v16,
            float* __restrict__ upd_acc, float* __restrict__ rowsum) {
  const int b     = blockIdx.y;
  const int chunk = blockIdx.x;               // 16 chunks of 256 tokens
  const int lane  = threadIdx.x & 31;
  const int wave  = threadIdx.x >> 5;         // 0..7
  __shared__ __align__(16) _Float16 q_lds[16 * 264];
  __shared__ __align__(16) _Float16 att[16 * 264];

  // q rows 0..7, zero-padded rows 8..15 (so padded M rows contribute nothing)
  for (int i = threadIdx.x; i < 16 * 256; i += 256) {
    const int r = i >> 8, c = i & 255;
    q_lds[r * 264 + c] = (r < NSLOT) ? q16[((long)b * NSLOT + r) * D + c] : (_Float16)0.f;
  }
  __syncthreads();

  const int tb = chunk * 256 + wave * 32;     // this wave's 32 tokens
  const _Float16* ktb = kt + (long)b * D * NTOK;
  v8f rs = {};
  #pragma unroll
  for (int t = 0; t < 2; ++t) {
    v8f acc = {};
    const _Float16* kbase = ktb + tb + t * 16;
    #pragma unroll
    for (int ks = 0; ks < 8; ++ks) {
      v16h A  = load_A(q_lds + ks * 32, 264, lane);
      v16h Bm = load_B(kbase + (long)(ks * 32) * NTOK, NTOK, lane);
      acc = wmma_f16(A, Bm, acc);             // dots tile (scale folded into q)
    }
    const int col = wave * 32 + t * 16 + (lane & 15);
    if (lane < 16) {
      // slots 0..7 for this token all live in this lane's 8 accumulators
      float mx = acc[0];
      #pragma unroll
      for (int r = 1; r < 8; ++r) mx = fmaxf(mx, acc[r]);
      float e[8]; float sum = 0.f;
      #pragma unroll
      for (int r = 0; r < 8; ++r) { e[r] = __expf(acc[r] - mx); sum += e[r]; }
      const float invs = 1.f / sum;
      #pragma unroll
      for (int r = 0; r < 8; ++r) {
        const float av = e[r] * invs + 1e-8f; // softmax + EPS
        rs[r] += av;                          // f32 row-sum for token renorm
        att[r * 264 + col] = (_Float16)av;
      }
    } else {
      #pragma unroll
      for (int r = 0; r < 8; ++r) att[(NSLOT + r) * 264 + col] = (_Float16)0.f;
    }
  }
  // reduce per-slot attn sums across the wave's 32 tokens, one atomic per slot
  #pragma unroll
  for (int r = 0; r < 8; ++r)
    #pragma unroll
    for (int m = 16; m; m >>= 1) rs[r] += __shfl_xor(rs[r], m, 32);
  if (lane == 0) {
    #pragma unroll
    for (int r = 0; r < 8; ++r) atomicAdd(&rowsum[b * NSLOT + r], rs[r]);
  }
  __syncthreads();

  // updates: contraction over the chunk's 256 tokens; wave owns 2 d-tiles
  const _Float16* vb = v16 + ((long)b * NTOK + chunk * 256) * D;
  #pragma unroll
  for (int t = 0; t < 2; ++t) {
    const int nt = wave * 2 + t;
    v8f acc = {};
    #pragma unroll
    for (int ks = 0; ks < 8; ++ks) {
      v16h A  = load_A(att + ks * 32, 264, lane);
      v16h Bm = load_B(vb + (long)(ks * 32) * D + nt * 16, D, lane);
      acc = wmma_f16(A, Bm, acc);
    }
    if (lane < 16) {                          // rows 0..7 = the real slots
      float* dst = upd_acc + (long)b * NSLOT * D + nt * 16 + lane;
      #pragma unroll
      for (int r = 0; r < 8; ++r) atomicAdd(dst + r * D, acc[r]);
    }
  }
}

// ---------------------------------------------------------------------------
// Finish renorm, GRU cell (torch r,z,n gate order), LayerNorm + residual MLP.
// 256 rows total -> VALU; one block per batch, thread = output column.
// ---------------------------------------------------------------------------
__global__ void __launch_bounds__(256)
slot_update_kernel(float* __restrict__ slots,
                   const float* __restrict__ upd_acc, const float* __restrict__ rowsum,
                   const float* __restrict__ gwi, const float* __restrict__ gwh,
                   const float* __restrict__ gbi, const float* __restrict__ gbh,
                   const float* __restrict__ lng, const float* __restrict__ lnb,
                   const float* __restrict__ w1, const float* __restrict__ b1,
                   const float* __restrict__ w2, const float* __restrict__ b2,
                   float* __restrict__ out, int write_out) {
  const int b = blockIdx.x;
  const int t = threadIdx.x;
  __shared__ float upd[NSLOT * D];
  __shared__ float sp[NSLOT * D];
  __shared__ float snew[NSLOT * D];
  __shared__ float hbuf[NSLOT * D];
  __shared__ float abuf[NSLOT * D];
  for (int i = t; i < NSLOT * D; i += 256) {
    upd[i] = upd_acc[(long)b * NSLOT * D + i] / rowsum[b * NSLOT + (i >> 8)];
    sp[i]  = slots[(long)b * NSLOT * D + i];
  }
  __syncthreads();

  float xr[NSLOT], xz[NSLOT], xn[NSLOT], hr[NSLOT], hz[NSLOT], hn[NSLOT];
  {
    const float bir = gbi[t], biz = gbi[D + t], bin = gbi[2 * D + t];
    const float bhr = gbh[t], bhz = gbh[D + t], bhn = gbh[2 * D + t];
    #pragma unroll
    for (int s = 0; s < NSLOT; ++s) {
      xr[s] = bir; xz[s] = biz; xn[s] = bin;
      hr[s] = bhr; hz[s] = bhz; hn[s] = bhn;
    }
  }
  for (int i = 0; i < D; ++i) {
    const float wir = gwi[i * 3 * D + t];
    const float wiz = gwi[i * 3 * D + D + t];
    const float win = gwi[i * 3 * D + 2 * D + t];
    const float whr = gwh[i * 3 * D + t];
    const float whz = gwh[i * 3 * D + D + t];
    const float whn = gwh[i * 3 * D + 2 * D + t];
    #pragma unroll
    for (int s = 0; s < NSLOT; ++s) {
      const float u = upd[s * D + i], h = sp[s * D + i];
      xr[s] = fmaf(u, wir, xr[s]); xz[s] = fmaf(u, wiz, xz[s]); xn[s] = fmaf(u, win, xn[s]);
      hr[s] = fmaf(h, whr, hr[s]); hz[s] = fmaf(h, whz, hz[s]); hn[s] = fmaf(h, whn, hn[s]);
    }
  }
  #pragma unroll
  for (int s = 0; s < NSLOT; ++s) {
    const float r  = 1.f / (1.f + __expf(-(xr[s] + hr[s])));
    const float z  = 1.f / (1.f + __expf(-(xz[s] + hz[s])));
    const float ng = tanhf(xn[s] + r * hn[s]);
    snew[s * D + t] = (1.f - z) * ng + z * sp[s * D + t];
  }
  __syncthreads();

  { // LayerNorm per slot row: one wave per row
    const int lane = t & 31, wv = t >> 5;
    float s1 = 0.f, s2 = 0.f, xv[8];
    #pragma unroll
    for (int e = 0; e < 8; ++e) {
      const float x = snew[wv * D + lane * 8 + e];
      xv[e] = x; s1 += x; s2 += x * x;
    }
    #pragma unroll
    for (int m = 16; m; m >>= 1) { s1 += __shfl_xor(s1, m, 32); s2 += __shfl_xor(s2, m, 32); }
    const float mean = s1 * (1.f / 256.f);
    const float inv  = rsqrtf(s2 * (1.f / 256.f) - mean * mean + 1e-5f);
    #pragma unroll
    for (int e = 0; e < 8; ++e) {
      const int c = lane * 8 + e;
      hbuf[wv * D + c] = (xv[e] - mean) * inv * lng[c] + lnb[c];
    }
  }
  __syncthreads();

  { // MLP layer 1 + relu
    float a[NSLOT];
    const float bb = b1[t];
    #pragma unroll
    for (int s = 0; s < NSLOT; ++s) a[s] = bb;
    for (int i = 0; i < D; ++i) {
      const float w = w1[i * D + t];
      #pragma unroll
      for (int s = 0; s < NSLOT; ++s) a[s] = fmaf(hbuf[s * D + i], w, a[s]);
    }
    #pragma unroll
    for (int s = 0; s < NSLOT; ++s) abuf[s * D + t] = fmaxf(a[s], 0.f);
  }
  __syncthreads();

  { // MLP layer 2 + residual, store
    float o[NSLOT];
    const float bb = b2[t];
    #pragma unroll
    for (int s = 0; s < NSLOT; ++s) o[s] = bb;
    for (int i = 0; i < D; ++i) {
      const float w = w2[i * D + t];
      #pragma unroll
      for (int s = 0; s < NSLOT; ++s) o[s] = fmaf(abuf[s * D + i], w, o[s]);
    }
    #pragma unroll
    for (int s = 0; s < NSLOT; ++s) {
      const float res = snew[s * D + t] + o[s];
      slots[(long)b * NSLOT * D + s * D + t] = res;
      if (write_out) out[(long)b * NSLOT * D + s * D + t] = res;
    }
  }
}

// ---------------------------------------------------------------------------
extern "C" void kernel_launch(void* const* d_in, const int* in_sizes, int n_in,
                              void* d_out, int out_size, void* d_ws, size_t ws_size,
                              hipStream_t stream) {
  const float* inputs     = (const float*)d_in[0];
  const float* ln_in_g    = (const float*)d_in[1];
  const float* ln_in_b    = (const float*)d_in[2];
  const float* ln_slots_g = (const float*)d_in[3];
  const float* ln_slots_b = (const float*)d_in[4];
  const float* ln_ff_g    = (const float*)d_in[5];
  const float* ln_ff_b    = (const float*)d_in[6];
  const float* slots_emb  = (const float*)d_in[7];
  const float* Wq = (const float*)d_in[8];   const float* bq = (const float*)d_in[9];
  const float* Wk = (const float*)d_in[10];  const float* bk = (const float*)d_in[11];
  const float* Wv = (const float*)d_in[12];  const float* bv = (const float*)d_in[13];
  const float* gru_wi = (const float*)d_in[14];
  const float* gru_wh = (const float*)d_in[15];
  const float* gru_bi = (const float*)d_in[16];
  const float* gru_bh = (const float*)d_in[17];
  const float* mlp_w1 = (const float*)d_in[18];
  const float* mlp_b1 = (const float*)d_in[19];
  const float* mlp_w2 = (const float*)d_in[20];
  const float* mlp_b2 = (const float*)d_in[21];

  char* ws = (char*)d_ws;
  size_t off = 0;
  _Float16* kt    = (_Float16*)(ws + off); off += (size_t)NB * D * NTOK * 2;  // 64 MB, K^T f16
  _Float16* v16   = (_Float16*)(ws + off); off += (size_t)NB * NTOK * D * 2;  // 64 MB, V f16
  _Float16* wk16  = (_Float16*)(ws + off); off += (size_t)D * D * 2;
  _Float16* wv16  = (_Float16*)(ws + off); off += (size_t)D * D * 2;
  _Float16* q16   = (_Float16*)(ws + off); off += (size_t)NB * NSLOT * D * 2;
  float*    slots = (float*)(ws + off);    off += (size_t)NB * NSLOT * D * 4;
  float*    upd   = (float*)(ws + off);    off += (size_t)NB * NSLOT * D * 4; // + rowsum below: contiguous
  float*    rsum  = (float*)(ws + off);    off += (size_t)NB * NSLOT * 4;

  convert_w_kernel<<<(D * D + 255) / 256, 256, 0, stream>>>(Wk, Wv, wk16, wv16);
  kv_proj_kernel<<<(NB * NTOK) / 64, 128, 0, stream>>>(inputs, ln_in_g, ln_in_b,
                                                       wk16, wv16, bk, bv, kt, v16);
  init_slots_kernel<<<NB, 256, 0, stream>>>(slots_emb, slots);

  for (int it = 0; it < N_ITERS; ++it) {
    q_proj_kernel<<<NB, 256, 0, stream>>>(slots, ln_slots_g, ln_slots_b, Wq, bq, q16);
    hipMemsetAsync(upd, 0, (size_t)NB * NSLOT * D * 4 + (size_t)NB * NSLOT * 4, stream);
    attn_kernel<<<dim3(NTOK / 256, NB), 256, 0, stream>>>(q16, kt, v16, upd, rsum);
    slot_update_kernel<<<NB, 256, 0, stream>>>(slots, upd, rsum,
        gru_wi, gru_wh, gru_bi, gru_bh, ln_ff_g, ln_ff_b,
        mlp_w1, mlp_b1, mlp_w2, mlp_b2, (float*)d_out, (it == N_ITERS - 1) ? 1 : 0);
  }
}
